// MultiHeadLocalAttention_27685359190795
// MI455X (gfx1250) — matrix operand order repaired and checked by reading, not compiled, not run on hardware
//
#include <hip/hip_runtime.h>
#include <hip/hip_bf16.h>
#include <stdint.h>

// MultiHeadLocalAttention (faithful torch-view semantics) for MI455X/gfx1250.
// HBM-bound (~1.05 GiB compulsory @ 23.3 TB/s ~= 45 us); fused per-(b,h,t)
// slab projection keeps the 1 GiB kproj/vproj intermediates out of HBM.
//  - GEMMs: v_wmma_f32_16x16x32_bf16 (fp32 accumulate)
//  - Raw k/v tiles staged into LDS by the Tensor Data Mover
//    (tensor_load_to_lds + s_wait_tensorcnt) with TDM padding (+1 DWORD per
//    256) to kill LDS bank conflicts; LDS region reused for the projected slab.
//  - TWO adjacent slabs per block (64x1024, single 2-D TDM descriptor):
//    halves per-block weight re-reads from L2 (the true bottleneck: weight
//    traffic drops 16 GiB -> 8 GiB).
//  - Weights pre-converted once to bf16; aout produced in bf16 so the final
//    Linear consumes it cvt-free.

#define B_  2
#define S_  2048
#define W_  32
#define D_  1024
#define H_  16
#define HD_ 64
#define T_  128   // S_/16

typedef __bf16 bf16_t;
typedef __attribute__((ext_vector_type(16))) __bf16 v16bf;
typedef __attribute__((ext_vector_type(8)))  float  v8f;
typedef uint32_t u32;
typedef __attribute__((ext_vector_type(4))) u32 u32x4;
typedef __attribute__((ext_vector_type(8))) u32 u32x8;

// TDM pad layout: +1 DWORD after every 256 DWORDs of the flat 64x1024 tile.
// Element (r,c) lives at r*1028 + c + (c>>8).
#define SLAB_LD     1028
#define SLAB_ROWS   64
#define SLAB_ELEMS  (SLAB_ROWS * SLAB_LD)     // 65792 dwords = 257 KB
#define SLAB_IDX(r, c) ((r) * SLAB_LD + (c) + ((c) >> 8))

static __device__ __forceinline__ v8f wmma_bf16(v16bf a, v16bf b, v8f c) {
  return __builtin_amdgcn_wmma_f32_16x16x32_bf16(false, a, false, b, (short)0, c,
                                                 false, false);
}

// ---------------------------------------------------------------------------
// TDM: DMA one 64x1024 fp32 tile (256 KB) global -> LDS as a 2-D tensor
// (32768 x 2 DWORDs), +1-DWORD padding every 256 DWORDs.
// ---------------------------------------------------------------------------
static __device__ __forceinline__ void tdm_load_slab(const float* gsrc,
                                                     float* smem) {
  const u32      lds_off = (u32)(uintptr_t)smem;  // low 32b of generic = LDS offset
  const uint64_t ga      = (uint64_t)(uintptr_t)gsrc;
  u32x4 g0;
  g0[0] = 1u;                                         // count=1 user descriptor
  g0[1] = lds_off;                                    // lds_addr
  g0[2] = (u32)ga;                                    // global_addr[31:0]
  g0[3] = (u32)((ga >> 32) & 0x01FFFFFFu) | (2u << 30);  // addr[56:32] | type=2
  u32x8 g1;
  g1[0] = (2u << 16)       // data_size = 4 bytes
        | (1u << 20)       // pad_enable
        | (7u << 22)       // pad_interval: every 256 DWORDs
        | (0u << 25);      // pad_amount: 1 DWORD
  g1[1] = 0x8000u << 16;   // tensor_dim0 = 32768 (low 16 bits at bit 48)
  g1[2] = 2u << 16;        // tensor_dim0 hi = 0 | tensor_dim1 = 2
  g1[3] = 0x8000u << 16;   // tensor_dim1 hi = 0 | tile_dim0 = 32768
  g1[4] = 2u;              // tile_dim1 = 2 | tile_dim2 = 0
  g1[5] = 32768u;          // tensor_dim0_stride low
  g1[6] = 0u;
  g1[7] = 0u;
  asm volatile("tensor_load_to_lds %0, %1" : : "s"(g0), "s"(g1) : "memory");
}

// A fragment (bf16 16x32 WMMA layout) from the padded fp32 LDS tile.
static __device__ __forceinline__ v16bf load_a_frag_lds(const float* sm,
                                                        int m0, int k0) {
  const int lane = threadIdx.x & 31;
  const int half = lane >> 4;
  const int l    = lane & 15;
  const int r    = m0 + l;
  const int c0   = k0 + 8 * half;       // 8-elem runs never straddle a pad
  const int c1   = c0 + 16;
  const float* p0 = sm + SLAB_IDX(r, c0);
  const float* p1 = sm + SLAB_IDX(r, c1);
  v16bf f;
#pragma unroll
  for (int i = 0; i < 8; ++i) f[i] = (bf16_t)p0[i];
#pragma unroll
  for (int i = 0; i < 8; ++i) f[i + 8] = (bf16_t)p1[i];
  return f;
}

// A fragment from row-major global, fp32 (cvt) or bf16 (direct) overloads.
static __device__ __forceinline__ v16bf load_a_frag_g(const float* __restrict__ A,
                                                      int lda, int m0, int k0) {
  const int lane = threadIdx.x & 31;
  const int half = lane >> 4;
  const int l    = lane & 15;
  const float* p = A + (size_t)(m0 + l) * lda + (k0 + 8 * half);
  v16bf f;
#pragma unroll
  for (int i = 0; i < 8; ++i) f[i] = (bf16_t)p[i];
#pragma unroll
  for (int i = 0; i < 8; ++i) f[i + 8] = (bf16_t)p[i + 16];
  return f;
}
static __device__ __forceinline__ v16bf load_a_frag_g(const bf16_t* __restrict__ A,
                                                      int lda, int m0, int k0) {
  const int lane = threadIdx.x & 31;
  const int half = lane >> 4;
  const int l    = lane & 15;
  const bf16_t* p = A + (size_t)(m0 + l) * lda + (k0 + 8 * half);
  v16bf f;
#pragma unroll
  for (int i = 0; i < 8; ++i) f[i] = p[i];
#pragma unroll
  for (int i = 0; i < 8; ++i) f[i + 8] = p[i + 16];
  return f;
}

// B fragment for B[k][n] = Wt[n][k], Wt bf16 [N][K] row-major.
// Per lane: two 16 B runs (k0+8*half, +16) -> two b128 loads, no cvt.
static __device__ __forceinline__ v16bf load_b_frag_bf(const bf16_t* __restrict__ Wbf,
                                                       int ldw, int n0, int k0) {
  const int lane = threadIdx.x & 31;
  const int half = lane >> 4;
  const int l    = lane & 15;
  const bf16_t* p = Wbf + (size_t)(n0 + l) * ldw + (k0 + 8 * half);
  v16bf f;
#pragma unroll
  for (int i = 0; i < 16; ++i) f[i] = p[(i & 7) + ((i >> 3) << 4)];
  return f;
}

// ---------------------------------------------------------------------------
// fp32 -> bf16 conversion (one-time weights).
// ---------------------------------------------------------------------------
__global__ __launch_bounds__(256) void cvt_bf16_kernel(const float* __restrict__ src,
                                                       bf16_t* __restrict__ dst,
                                                       int n) {
  const int i = blockIdx.x * 256 + threadIdx.x;
  if (i < n) dst[i] = (bf16_t)src[i];
}

// ---------------------------------------------------------------------------
// Generic Linear: C[M][N] = A[M][K] @ Wt^T + bias (Wt bf16 [N][K]).
// 256 threads (8 waves), block tile 128x128, wave tile 64x32.
// ---------------------------------------------------------------------------
template <typename AT>
__global__ __launch_bounds__(256) void linear_kernel(
    const AT* __restrict__ A, const bf16_t* __restrict__ Wbf,
    const float* __restrict__ bias, float* __restrict__ C,
    int M, int N, int K) {
  const int wave   = threadIdx.x >> 5;
  const int m_base = blockIdx.x * 128 + (wave >> 2) * 64;
  const int n_base = blockIdx.y * 128 + (wave & 3) * 32;

  v8f acc[4][2] = {};
  for (int k0 = 0; k0 < K; k0 += 32) {
    v16bf b0 = load_b_frag_bf(Wbf, K, n_base, k0);
    v16bf b1 = load_b_frag_bf(Wbf, K, n_base + 16, k0);
#pragma unroll
    for (int mi = 0; mi < 4; ++mi) {
      v16bf a = load_a_frag_g(A, K, m_base + mi * 16, k0);
      acc[mi][0] = wmma_bf16(a, b0, acc[mi][0]);
      acc[mi][1] = wmma_bf16(a, b1, acc[mi][1]);
    }
  }
  const int lane = threadIdx.x & 31, half = lane >> 4, l = lane & 15;
#pragma unroll
  for (int mi = 0; mi < 4; ++mi) {
#pragma unroll
    for (int ni = 0; ni < 2; ++ni) {
      const int   nn = n_base + ni * 16 + l;
      const float bb = bias[nn];
#pragma unroll
      for (int r = 0; r < 8; ++r) {
        const int mm = m_base + mi * 16 + r + 8 * half;
        C[(size_t)mm * N + nn] = acc[mi][ni][r] + bb;
      }
    }
  }
}

// ---------------------------------------------------------------------------
// Project a 64x1024 double-slab (M=64,N=1024,K=1024) into LDS, in place.
// 512 threads (16 waves); wave owns 64 output columns -> acc[4 Mtiles][4 Ntiles].
// ---------------------------------------------------------------------------
static __device__ __forceinline__ void project_slab(
    const float* __restrict__ abase,      // 64x1024 fp32 rows, contiguous
    const bf16_t* __restrict__ Wbf,       // [1024][1024] bf16
    const float* __restrict__ bias,       // [1024] fp32
    float* __restrict__ smem) {           // SLAB_ELEMS floats
  if (threadIdx.x < 32) {
    tdm_load_slab(abase, smem);
    __builtin_amdgcn_s_wait_tensorcnt(0);
  }
  __syncthreads();                        // raw A visible to all waves

  const int wave   = threadIdx.x >> 5;    // 0..15
  const int n_wave = wave * 64;
  v8f acc[4][4] = {};
  for (int k0 = 0; k0 < D_; k0 += 32) {
    v16bf a[4];
#pragma unroll
    for (int mi = 0; mi < 4; ++mi) a[mi] = load_a_frag_lds(smem, mi * 16, k0);
#pragma unroll
    for (int ni = 0; ni < 4; ++ni) {
      v16bf bf = load_b_frag_bf(Wbf, D_, n_wave + ni * 16, k0);
#pragma unroll
      for (int mi = 0; mi < 4; ++mi)
        acc[mi][ni] = wmma_bf16(a[mi], bf, acc[mi][ni]);
    }
  }
  __syncthreads();                        // everyone done reading raw A

  const int lane = threadIdx.x & 31, half = lane >> 4, l = lane & 15;
#pragma unroll
  for (int mi = 0; mi < 4; ++mi) {
#pragma unroll
    for (int ni = 0; ni < 4; ++ni) {
      const int   cc = n_wave + ni * 16 + l;
      const float bb = bias[cc];
#pragma unroll
      for (int r = 0; r < 8; ++r)
        smem[SLAB_IDX(mi * 16 + r + 8 * half, cc)] = acc[mi][ni][r] + bb;
    }
  }
  __syncthreads();                        // slab visible for the epilogue
}

// ---------------------------------------------------------------------------
// Scores: block per (b,h,r) with t = 2r+tp, tp in {0,1}; s_src = h*128+2r.
// scores[b,h,32r+jj,w] = (1/8) sum_e qproj[b,32r+jj,h*64+e] *
//                          slab[(jj>>4)*32 + 2*(jj&15) + (w>>4)][(w&15)*64+e]
// stored [b][s][w][h] so the head-axis softmax reduction is contiguous.
// ---------------------------------------------------------------------------
__global__ __launch_bounds__(512) void slab_scores_kernel(
    const float* __restrict__ k, const bf16_t* __restrict__ Wkbf,
    const float* __restrict__ bk, const float* __restrict__ qproj,
    float* __restrict__ scores) {
  const int blk = blockIdx.x;             // b*(H*64) + h*64 + r
  const int r = blk & 63;
  const int h = (blk >> 6) & (H_ - 1);
  const int b = blk >> 10;
  const int s_src = h * T_ + 2 * r;

  __shared__ float smem[SLAB_ELEMS];      // raw A, then projected slab
  __shared__ float qtile[32][HD_];

  const float* qb = qproj + (size_t)(b * S_ + r * 32) * D_ + h * HD_;
  for (int i = threadIdx.x; i < 32 * HD_; i += 512)
    qtile[i >> 6][i & 63] = qb[(size_t)(i >> 6) * D_ + (i & 63)];

  project_slab(k + (size_t)(b * S_ + s_src) * W_ * D_, Wkbf, bk, smem);

  for (int idx = threadIdx.x; idx < 1024; idx += 512) {
    const int jj = idx >> 5, w = idx & 31;
    const int tp = jj >> 4, j = jj & 15;
    const int u = w >> 4, m = w & 15;
    const int c0 = m * 64;                // 64-runs never straddle a pad
    const float* sr = smem + SLAB_IDX(tp * 32 + 2 * j + u, c0);
    float s = 0.f;
#pragma unroll
    for (int e = 0; e < 64; ++e) s += qtile[jj][e] * sr[e];
    scores[((size_t)((b * S_ + r * 32 + jj) * W_ + w)) * H_ + h] = s * 0.125f;
  }
}

// ---------------------------------------------------------------------------
// Softmax normalizers over the HEAD axis: one thread per (b,s,w).
// ---------------------------------------------------------------------------
__global__ __launch_bounds__(256) void z_kernel(
    const float* __restrict__ scores, float* __restrict__ zmax,
    float* __restrict__ zsum, int n) {
  const int i = blockIdx.x * 256 + threadIdx.x;
  if (i >= n) return;
  const float* p = scores + (size_t)i * H_;
  float mx = p[0];
#pragma unroll
  for (int h = 1; h < H_; ++h) mx = fmaxf(mx, p[h]);
  float s = 0.f;
#pragma unroll
  for (int h = 0; h < H_; ++h) s += __expf(p[h] - mx);
  zmax[i] = mx;
  zsum[i] = s;
}

// ---------------------------------------------------------------------------
// Attention-apply: block per (b,h,r); emits TWO contiguous pre-Wo rows in bf16:
// aout[b][h*128+2r+tp][j*64+e] = sum_w attn*slab[tp*32+2j+(w>>4)][(w&15)*64+e]
// ---------------------------------------------------------------------------
__global__ __launch_bounds__(512) void slab_attn_kernel(
    const float* __restrict__ v, const bf16_t* __restrict__ Wvbf,
    const float* __restrict__ bv, const float* __restrict__ scores,
    const float* __restrict__ zmax, const float* __restrict__ zsum,
    bf16_t* __restrict__ aout) {
  const int blk = blockIdx.x;
  const int r = blk & 63;
  const int h = (blk >> 6) & (H_ - 1);
  const int b = blk >> 10;
  const int s_src = h * T_ + 2 * r;

  __shared__ float smem[SLAB_ELEMS];
  __shared__ float attnw[32][W_];

  for (int idx = threadIdx.x; idx < 1024; idx += 512) {
    const int jj = idx >> 5, w = idx & 31;
    const size_t row = (size_t)(b * S_ + r * 32 + jj) * W_ + w;
    const float sc = scores[row * H_ + h];
    attnw[jj][w] = __expf(sc - zmax[row]) / zsum[row];
  }

  project_slab(v + (size_t)(b * S_ + s_src) * W_ * D_, Wvbf, bv, smem);

  for (int idx = threadIdx.x; idx < 2048; idx += 512) {
    const int jj = idx >> 6, e = idx & 63;
    const int tp = jj >> 4, j = jj & 15;
    float s = 0.f;
#pragma unroll
    for (int w = 0; w < W_; ++w) {
      const int c = (w & 15) * 64 + e;
      s += attnw[jj][w] * smem[SLAB_IDX(tp * 32 + 2 * j + (w >> 4), c)];
    }
    aout[(size_t)(b * S_ + h * T_ + 2 * r + tp) * D_ + j * 64 + e] = (bf16_t)s;
  }
}

// ---------------------------------------------------------------------------
extern "C" void kernel_launch(void* const* d_in, const int* in_sizes, int n_in,
                              void* d_out, int out_size, void* d_ws, size_t ws_size,
                              hipStream_t stream) {
  (void)in_sizes; (void)n_in; (void)out_size; (void)ws_size;
  const float* q  = (const float*)d_in[0];
  const float* k  = (const float*)d_in[1];
  const float* v  = (const float*)d_in[2];
  const float* Wq = (const float*)d_in[3];
  const float* bq = (const float*)d_in[4];
  const float* Wk = (const float*)d_in[5];
  const float* bk = (const float*)d_in[6];
  const float* Wv = (const float*)d_in[7];
  const float* bv = (const float*)d_in[8];
  const float* Wo = (const float*)d_in[9];
  const float* bo = (const float*)d_in[10];
  float* out = (float*)d_out;

  float* ws     = (float*)d_ws;
  float* qproj  = ws;                                   // B*S*D   = 4194304 f32
  float* scores = qproj + (size_t)B_ * S_ * D_;         // B*S*W*H = 2097152 f32
  float* zmax   = scores + (size_t)B_ * S_ * W_ * H_;   // B*S*W   =  131072 f32
  float* zsum   = zmax + (size_t)B_ * S_ * W_;          // B*S*W   =  131072 f32
  bf16_t* aoutb = (bf16_t*)(zsum + (size_t)B_ * S_ * W_);  // B*S*D bf16
  bf16_t* wqbf  = aoutb + (size_t)B_ * S_ * D_;         // 4 x 1M bf16
  bf16_t* wkbf  = wqbf + (size_t)D_ * D_;
  bf16_t* wvbf  = wkbf + (size_t)D_ * D_;
  bf16_t* wobf  = wvbf + (size_t)D_ * D_;

  const dim3 b256(256), b512(512);
  const int  nw = D_ * D_;
  // 0) one-time weight conversion to bf16
  cvt_bf16_kernel<<<dim3(nw / 256), b256, 0, stream>>>(Wq, wqbf, nw);
  cvt_bf16_kernel<<<dim3(nw / 256), b256, 0, stream>>>(Wk, wkbf, nw);
  cvt_bf16_kernel<<<dim3(nw / 256), b256, 0, stream>>>(Wv, wvbf, nw);
  cvt_bf16_kernel<<<dim3(nw / 256), b256, 0, stream>>>(Wo, wobf, nw);
  // 1) qproj = q @ Wq^T + bq
  linear_kernel<float><<<dim3((B_ * S_) / 128, D_ / 128), b256, 0, stream>>>(
      q, wqbf, bq, qproj, B_ * S_, D_, D_);
  // 2) fused k-projection + scores (two slabs per block)
  slab_scores_kernel<<<dim3(B_ * H_ * 64), b512, 0, stream>>>(
      k, wkbf, bk, qproj, scores);
  // 3) head-axis softmax normalizers
  z_kernel<<<dim3((B_ * S_ * W_) / 256), b256, 0, stream>>>(
      scores, zmax, zsum, B_ * S_ * W_);
  // 4) fused v-projection + attention apply (bf16 output)
  slab_attn_kernel<<<dim3(B_ * H_ * 64), b512, 0, stream>>>(
      v, wvbf, bv, scores, zmax, zsum, aoutb);
  // 5) out = aout @ Wo^T + bo
  linear_kernel<bf16_t><<<dim3((B_ * S_) / 128, D_ / 128), b256, 0, stream>>>(
      aoutb, wobf, bo, out, B_ * S_, D_, D_);
}
